// FullAttention_12627203850747
// MI455X (gfx1250) — matrix operand order repaired and checked by reading, not compiled, not run on hardware
//
#include <hip/hip_runtime.h>
#include <hip/hip_bf16.h>
#include <math.h>

typedef __attribute__((ext_vector_type(16))) _Float16 v16h;
typedef __attribute__((ext_vector_type(8)))  float    v8f;

#define B_  2
#define L_  2048
#define H_  16
#define E_  64
#define D_  64
#define BQ  128            // query rows per block (8 waves x 16)
#define BK  64             // kv rows per tile
#define LSTR 80            // LDS row stride in halfs (16B aligned rows)

// ds_swizzle_b32 XOR butterfly within 16-lane groups (masks 1..8 never cross lane 16)
#define SWZ_MAX(x, m)                                                          \
    (x) = fmaxf((x), __int_as_float(__builtin_amdgcn_ds_swizzle(               \
                         __float_as_int(x), ((m) << 10) | 0x1f)))

union U16 { v16h v; _Float16 h[16]; int4 q[2]; };

// ---------------- suffix-sum pre-pass: ws[bh][l][d] = sum_{s>l} V[b][s][h][d]
__global__ void suffix_kernel(const float* __restrict__ V, float* __restrict__ ws) {
    const int bh = blockIdx.x;           // 0..31
    const int b  = bh >> 4;
    const int h  = bh & 15;
    const int d  = threadIdx.x;          // 0..63
    float acc = 0.0f;
    for (int s = L_ - 1; s >= 0; --s) {
        ws[((size_t)bh * L_ + s) * D_ + d] = acc;
        acc += V[(((size_t)b * L_ + s) * H_ + h) * D_ + d];
    }
}

// ---------------- flash attention with multiplicative-mask correction
__global__ void __launch_bounds__(256)
attn_kernel(const float* __restrict__ Qg, const float* __restrict__ Kg,
            const float* __restrict__ Vg, const float* __restrict__ ws,
            float* __restrict__ Og) {
    __shared__ _Float16 Ksh[2][BK * LSTR];     // [buf][s][e]
    __shared__ _Float16 Vtsh[2][D_ * LSTR];    // [buf][d][s] (transposed V tile)
    __shared__ _Float16 Psh[8 * 16 * LSTR];    // per-wave P scratch [row][s]

    const int tid  = threadIdx.x;
    const int lane = tid & 31;
    const int wv   = tid >> 5;                 // 0..7
    const int col  = lane & 15;
    const int hi   = (lane >> 4) & 1;          // 0: lanes 0-15, 1: lanes 16-31
    const int roff = hi ? 8 : 0;               // row offset for C-layout / stats

    const int bh = blockIdx.y;
    const int b  = bh >> 4;
    const int h  = bh & 15;
    const int q0 = blockIdx.x * BQ;
    const int r0 = q0 + wv * 16;               // this wave's first query row

    const float scale = 0.125f;                // 1/sqrt(64), folded into Q

    // staging thread mapping (shared by all tiles)
    const int ksrow = tid >> 2;                // 0..63
    const int kec   = (tid & 3) * 16;          // e chunk
    const int vsrow = tid & 63;                // 0..63
    const int vdc   = (tid >> 6) * 16;         // d chunk
    const float* kbase = Kg + (((size_t)b * L_ + ksrow) * H_ + h) * E_ + kec;
    const float* vbase = Vg + (((size_t)b * L_ + vsrow) * H_ + h) * D_ + vdc;

    float4 kf[4], vf[4];                       // in-flight staged tile
    auto loadKV = [&](int s0) {
#pragma unroll
        for (int i = 0; i < 4; ++i) {
            kf[i] = *(const float4*)(kbase + (size_t)s0 * H_ * E_ + i * 4);
            vf[i] = *(const float4*)(vbase + (size_t)s0 * H_ * D_ + i * 4);
        }
    };
    auto storeKV = [&](_Float16* Kbuf, _Float16* Vbuf) {
        U16 u;
#pragma unroll
        for (int i = 0; i < 4; ++i) {
            u.h[4*i+0] = (_Float16)kf[i].x; u.h[4*i+1] = (_Float16)kf[i].y;
            u.h[4*i+2] = (_Float16)kf[i].z; u.h[4*i+3] = (_Float16)kf[i].w;
        }
        _Float16* dst = &Kbuf[ksrow * LSTR + kec];
        ((int4*)dst)[0] = u.q[0];
        ((int4*)dst)[1] = u.q[1];
#pragma unroll
        for (int i = 0; i < 4; ++i) {
            Vbuf[(vdc + 4*i + 0) * LSTR + vsrow] = (_Float16)vf[i].x;
            Vbuf[(vdc + 4*i + 1) * LSTR + vsrow] = (_Float16)vf[i].y;
            Vbuf[(vdc + 4*i + 2) * LSTR + vsrow] = (_Float16)vf[i].z;
            Vbuf[(vdc + 4*i + 3) * LSTR + vsrow] = (_Float16)vf[i].w;
        }
    };

    // ---- load Q fragments (A-matrix layout), pre-scaled by 1/sqrt(E)
    const int kbA = hi ? 16 : 0;
    U16 qa[2];
    {
        const float* qrow = Qg + (((size_t)b * L_ + (r0 + col)) * H_ + h) * E_;
#pragma unroll
        for (int c = 0; c < 2; ++c) {
            const int e0 = c * 32 + kbA;
            float4 f0 = *(const float4*)(qrow + e0);
            float4 f1 = *(const float4*)(qrow + e0 + 4);
            float4 f2 = *(const float4*)(qrow + e0 + 8);
            float4 f3 = *(const float4*)(qrow + e0 + 12);
            qa[c].h[0]  = (_Float16)(f0.x * scale); qa[c].h[1]  = (_Float16)(f0.y * scale);
            qa[c].h[2]  = (_Float16)(f0.z * scale); qa[c].h[3]  = (_Float16)(f0.w * scale);
            qa[c].h[4]  = (_Float16)(f1.x * scale); qa[c].h[5]  = (_Float16)(f1.y * scale);
            qa[c].h[6]  = (_Float16)(f1.z * scale); qa[c].h[7]  = (_Float16)(f1.w * scale);
            qa[c].h[8]  = (_Float16)(f2.x * scale); qa[c].h[9]  = (_Float16)(f2.y * scale);
            qa[c].h[10] = (_Float16)(f2.z * scale); qa[c].h[11] = (_Float16)(f2.w * scale);
            qa[c].h[12] = (_Float16)(f3.x * scale); qa[c].h[13] = (_Float16)(f3.y * scale);
            qa[c].h[14] = (_Float16)(f3.z * scale); qa[c].h[15] = (_Float16)(f3.w * scale);
        }
    }

    // ---- all-ones B fragment: P x Ones accumulates row sums (softmax denom)
    U16 ones;
    {
        const int oh = 0x3C003C00;             // two f16 1.0
        int4 op; op.x = oh; op.y = oh; op.z = oh; op.w = oh;
        ones.q[0] = op; ones.q[1] = op;
    }

    // ---- accumulators (C-layout: VGPR v -> row v or 8+v; lane&15 -> col)
    v8f o[4], osum;
#pragma unroll
    for (int nbd = 0; nbd < 4; ++nbd)
#pragma unroll
        for (int i = 0; i < 8; ++i) o[nbd][i] = 0.0f;
#pragma unroll
    for (int i = 0; i < 8; ++i) osum[i] = 0.0f;
    float m_r[8];
#pragma unroll
    for (int v = 0; v < 8; ++v) m_r[v] = 0.0f;   // m init 0: masked zeros

    const int kb16 = hi ? 16 : 0;
    _Float16* Pw = &Psh[wv * 16 * LSTR];

    const int nkv = q0 / BK + 2;               // tiles covering s <= q0+BQ-1

    // prologue: stage tile 0
    loadKV(0);
    storeKV(Ksh[0], Vtsh[0]);

    for (int j = 0; j < nkv; ++j) {
        const int s0  = j * BK;
        const int cur = j & 1;
        const bool more = (j + 1 < nkv);
        __syncthreads();                       // tile j visible; prev reads done

        if (more) {
            loadKV(s0 + BK);                   // issue next tile's global loads
            if (j + 2 < nkv) {                 // warm L2 two tiles ahead
                __builtin_prefetch(kbase + (size_t)(s0 + 2 * BK) * H_ * E_, 0, 1);
                __builtin_prefetch(vbase + (size_t)(s0 + 2 * BK) * H_ * D_, 0, 1);
            }
        }

        if (s0 <= r0 + 15) {                   // tile not fully above diagonal
            const _Float16* Kc = Ksh[cur];
            const _Float16* Vc = Vtsh[cur];

            // ---- S = Q * K^T (4 column blocks x 2 k-chunks), pre-scaled
            v8f sreg[4];
#pragma unroll
            for (int nb = 0; nb < 4; ++nb) {
                v8f cs;
#pragma unroll
                for (int i = 0; i < 8; ++i) cs[i] = 0.0f;
#pragma unroll
                for (int c = 0; c < 2; ++c) {
                    U16 bk;
                    const _Float16* kp = &Kc[(nb * 16 + col) * LSTR + c * 32 + kb16];
                    bk.q[0] = ((const int4*)kp)[0];
                    bk.q[1] = ((const int4*)kp)[1];
                    cs = __builtin_amdgcn_wmma_f32_16x16x32_f16(
                            false, qa[c].v, false, bk.v, (short)0, cs, false, false);
                }
                sreg[nb] = cs;
            }

            // ---- online max + P write; denominator via P x Ones WMMA
            if (s0 + BK - 1 <= r0) {
                // fast path: fully below diagonal -> no masking
#pragma unroll
                for (int v = 0; v < 8; ++v) {
                    float mx = m_r[v];
#pragma unroll
                    for (int nb = 0; nb < 4; ++nb) mx = fmaxf(mx, sreg[nb][v]);
                    SWZ_MAX(mx, 1); SWZ_MAX(mx, 2); SWZ_MAX(mx, 4); SWZ_MAX(mx, 8);
                    const float alpha = __expf(m_r[v] - mx);
                    m_r[v] = mx;
                    const int prow = v + roff;
#pragma unroll
                    for (int nb = 0; nb < 4; ++nb)
                        Pw[prow * LSTR + nb * 16 + col] = (_Float16)__expf(sreg[nb][v] - mx);
                    osum[v] *= alpha;
#pragma unroll
                    for (int nbd = 0; nbd < 4; ++nbd) o[nbd][v] *= alpha;
                }
            } else {
                // diagonal tile: mask s > l (exp of -3e38 underflows to 0)
#pragma unroll
                for (int v = 0; v < 8; ++v) {
                    const int lrow = r0 + v + roff;
                    float vals[4];
                    float mx = m_r[v];
#pragma unroll
                    for (int nb = 0; nb < 4; ++nb) {
                        const int sg = s0 + nb * 16 + col;
                        vals[nb] = (sg <= lrow) ? sreg[nb][v] : -3.0e38f;
                        mx = fmaxf(mx, vals[nb]);
                    }
                    SWZ_MAX(mx, 1); SWZ_MAX(mx, 2); SWZ_MAX(mx, 4); SWZ_MAX(mx, 8);
                    const float alpha = __expf(m_r[v] - mx);
                    m_r[v] = mx;
                    const int prow = v + roff;
#pragma unroll
                    for (int nb = 0; nb < 4; ++nb)
                        Pw[prow * LSTR + nb * 16 + col] = (_Float16)__expf(vals[nb] - mx);
                    osum[v] *= alpha;
#pragma unroll
                    for (int nbd = 0; nbd < 4; ++nbd) o[nbd][v] *= alpha;
                }
            }

            // ---- reload P as A fragments (same-wave LDS, in-order)
            U16 pa[2];
#pragma unroll
            for (int c = 0; c < 2; ++c) {
                const _Float16* pp = &Pw[col * LSTR + c * 32 + kbA];
                pa[c].q[0] = ((const int4*)pp)[0];
                pa[c].q[1] = ((const int4*)pp)[1];
            }
            // ---- O += P * V ; denom += P * Ones
#pragma unroll
            for (int nbd = 0; nbd < 4; ++nbd) {
#pragma unroll
                for (int c = 0; c < 2; ++c) {
                    U16 bv;
                    const _Float16* vp = &Vc[(nbd * 16 + col) * LSTR + c * 32 + kb16];
                    bv.q[0] = ((const int4*)vp)[0];
                    bv.q[1] = ((const int4*)vp)[1];
                    o[nbd] = __builtin_amdgcn_wmma_f32_16x16x32_f16(
                                false, pa[c].v, false, bv.v, (short)0, o[nbd], false, false);
                }
            }
#pragma unroll
            for (int c = 0; c < 2; ++c)
                osum = __builtin_amdgcn_wmma_f32_16x16x32_f16(
                            false, pa[c].v, false, ones.v, (short)0, osum, false, false);
        }

        if (more)
            storeKV(Ksh[1 - cur], Vtsh[1 - cur]);   // stage tile j+1 into alt buffer
    }

    // ---- epilogue: add masked-slot (suffix) contribution and normalize
#pragma unroll
    for (int v = 0; v < 8; ++v) {
        const int lrow = r0 + v + roff;
        const float em = __expf(-m_r[v]);
        const float denom = osum[v] + em * (float)(L_ - 1 - lrow);
        const float inv = 1.0f / denom;
        const float* sfx = ws + ((size_t)bh * L_ + lrow) * D_;
        float* orow = Og + (((size_t)b * L_ + lrow) * H_ + h) * D_;
#pragma unroll
        for (int nbd = 0; nbd < 4; ++nbd) {
            const int d = nbd * 16 + col;
            orow[d] = (o[nbd][v] + em * sfx[d]) * inv;
        }
    }
}

extern "C" void kernel_launch(void* const* d_in, const int* in_sizes, int n_in,
                              void* d_out, int out_size, void* d_ws, size_t ws_size,
                              hipStream_t stream) {
    (void)in_sizes; (void)n_in; (void)out_size; (void)ws_size;
    const float* Qg = (const float*)d_in[0];
    const float* Kg = (const float*)d_in[1];
    const float* Vg = (const float*)d_in[2];
    float* Og = (float*)d_out;
    float* ws = (float*)d_ws;   // needs B*H*L*D*4 = 16 MB

    suffix_kernel<<<dim3(B_ * H_), dim3(D_), 0, stream>>>(Vg, ws);
    attn_kernel<<<dim3(L_ / BQ, B_ * H_), dim3(256), 0, stream>>>(Qg, Kg, Vg, ws, Og);
}
